// FSGNN_78254304133531
// MI455X (gfx1250) — compile-verified, matrix operand order checked by reading, not run on hardware
//
#include <hip/hip_runtime.h>

// Problem constants (fixed by the reference)
#define NN 50000
#define EE 600000
#define RR 8
#define BB 4
#define GG 64
#define CC 16
#define DD 128

typedef __attribute__((ext_vector_type(16))) _Float16 v16h;
typedef __attribute__((ext_vector_type(8)))  float    v8f;

// ---------- helpers ----------
__device__ inline v16h load_frag(const _Float16* p0, const _Float16* p1) {
    union { v16h v; uint4 q[2]; } u;
    u.q[0] = *(const uint4*)p0;
    u.q[1] = *(const uint4*)p1;
    return u.v;
}

// ---------- zero workspace ----------
__global__ void k_zero(unsigned* __restrict__ p, long long n) {
    long long i = (long long)blockIdx.x * blockDim.x + threadIdx.x;
    if (i < n) p[i] = 0u;
}

// ---------- build f16 weights in WMMA fragment order ----------
// Layout: Wh[ ((r*4 + kt)*8 + nt)*512 + lane*16 + elem ]
// relation r in [0,8]; r==8 is the root matrix.
__global__ void k_prepW(const float* __restrict__ basis, const float* __restrict__ comp,
                        const float* __restrict__ root, _Float16* __restrict__ Wh) {
    int idx = blockIdx.x * blockDim.x + threadIdx.x;
    if (idx >= 9 * 4 * 8 * 512) return;
    int elem = idx & 15;
    int lane = (idx >> 4) & 31;
    int nt   = (idx >> 9) & 7;
    int kt   = (idx >> 12) & 3;
    int r    =  idx >> 14;
    int m  = lane & 15;
    int hi = lane >> 4;
    int kloc = (elem & 7) + ((elem >> 3) << 4) + hi * 8;  // 0-7/16-23 (lo lanes), 8-15/24-31 (hi lanes)
    int kg  = kt * 32 + kloc;
    int col = nt * 16 + m;
    float v;
    if (r < RR) {
        v = 0.f;
        #pragma unroll
        for (int b = 0; b < BB; b++)
            v += comp[r * BB + b] * basis[((size_t)b * DD + kg) * DD + col];
    } else {
        v = root[(size_t)kg * DD + col];
    }
    Wh[idx] = (_Float16)v;
}

// ---------- convert x to f16, accumulate per-graph node counts ----------
__global__ void k_xh(const float* __restrict__ x, _Float16* __restrict__ xh,
                     const int* __restrict__ batch, float* __restrict__ gcnt) {
    long long idx = (long long)blockIdx.x * blockDim.x + threadIdx.x;
    if (idx >= (long long)NN * DD) return;
    xh[idx] = (_Float16)x[idx];
    if ((idx & (DD - 1)) == 0)
        atomicAdd(&gcnt[batch[idx >> 7]], 1.0f);
}

// ---------- per-(relation,dst) edge counts + per-relation totals ----------
// ctrl: [0..7]=relTotal  [8..15]=relCursor  [16..24]=edgeOff  [25..33]=tilePrefix
__global__ void k_count(const int* __restrict__ etype, const int* __restrict__ dst,
                        unsigned* __restrict__ cnt_rd, unsigned* __restrict__ ctrl) {
    int e = blockIdx.x * blockDim.x + threadIdx.x;
    if (e >= EE) return;
    int r = etype[e];
    int d = dst[e];
    atomicAdd(&cnt_rd[(size_t)r * NN + d], 1u);
    atomicAdd(&ctrl[r], 1u);
}

// ---------- prefix sums (1 thread) ----------
__global__ void k_offsets(unsigned* __restrict__ ctrl) {
    unsigned eo = 0, tp = 0;
    for (int r = 0; r < RR; r++) {
        ctrl[16 + r] = eo;
        ctrl[25 + r] = tp;
        unsigned c = ctrl[r];
        eo += c;
        tp += (c + 15u) >> 4;
    }
    ctrl[24] = eo;   // == EE
    ctrl[33] = tp;   // total tiles
}

// ---------- counting-sort edges by relation ----------
__global__ void k_scatter(const int* __restrict__ etype, unsigned* __restrict__ perm,
                          unsigned* __restrict__ ctrl) {
    int e = blockIdx.x * blockDim.x + threadIdx.x;
    if (e >= EE) return;
    int r = etype[e];
    unsigned pos = atomicAdd(&ctrl[8 + r], 1u);
    perm[ctrl[16 + r] + pos] = (unsigned)e;
}

// ---------- edge-message WMMA kernel ----------
// One wave per 16-edge tile (all same relation): msg = x[src] @ W[r], scaled by 1/cnt, atomic-add to agg.
__global__ __launch_bounds__(128) void k_msg(
    const _Float16* __restrict__ xh, const _Float16* __restrict__ Wh,
    const int* __restrict__ srcIdx, const int* __restrict__ dstIdx,
    const unsigned* __restrict__ perm, const unsigned* __restrict__ cnt_rd,
    const unsigned* __restrict__ ctrl, float* __restrict__ agg) {
    __shared__ int   s_src[4][16];
    __shared__ int   s_dst[4][16];
    __shared__ float s_scl[4][16];

    int w    = threadIdx.x >> 5;
    int lane = threadIdx.x & 31;
    int t    = blockIdx.x * 4 + w;

    const unsigned* eo = ctrl + 16;   // edgeOff[9]
    const unsigned* tp = ctrl + 25;   // tilePrefix[9]
    unsigned totTiles = tp[8];
    bool active = ((unsigned)t < totTiles);

    int r = 0, base = 0, cnt = 0;
    if (active) {
        while (r < RR - 1 && (unsigned)t >= tp[r + 1]) r++;
        int ti = t - (int)tp[r];
        base = (int)eo[r] + ti * 16;
        int rem = (int)eo[r + 1] - base;
        cnt = rem < 16 ? rem : 16;
    }
    if (lane < 16) {
        int srow = 0, d = 0;
        float sc = 0.f;
        if (active && lane < cnt) {
            int e = (int)perm[base + lane];
            srow = srcIdx[e];
            d = dstIdx[e];
            unsigned c = cnt_rd[(size_t)r * NN + d];
            sc = 1.0f / (float)(c ? c : 1u);
        }
        s_src[w][lane] = srow;
        s_dst[w][lane] = d;
        s_scl[w][lane] = sc;
    }
    __syncthreads();
    if (!active) return;

    int m  = lane & 15;
    int hi = lane >> 4;
    const _Float16* arow = xh + (size_t)s_src[w][m] * DD + hi * 8;
    const _Float16* wrel = Wh + (size_t)r * (4 * 8 * 512);

    v8f acc[8] = {};
    #pragma unroll
    for (int kt = 0; kt < 4; kt++) {
        v16h a = load_frag(arow + kt * 32, arow + kt * 32 + 16);
        const _Float16* wk = wrel + kt * (8 * 512);
        #pragma unroll
        for (int nt = 0; nt < 8; nt++) {
            const _Float16* bp = wk + nt * 512 + lane * 16;
            v16h b = load_frag(bp, bp + 8);
            acc[nt] = __builtin_amdgcn_wmma_f32_16x16x32_f16(
                false, a, false, b, (short)0, acc[nt], false, false);
        }
    }
    #pragma unroll
    for (int nt = 0; nt < 8; nt++) {
        int col = nt * 16 + m;
        #pragma unroll
        for (int i = 0; i < 8; i++) {
            int row = i + hi * 8;
            float sc = s_scl[w][row];
            if (sc != 0.f)
                atomicAdd(&agg[(size_t)s_dst[w][row] * DD + col], acc[nt][i] * sc);
        }
    }
}

// ---------- root GEMM + bias + agg + ReLU + pooled sum ----------
__global__ __launch_bounds__(128) void k_h(
    const _Float16* __restrict__ xh, const _Float16* __restrict__ Wh,
    const float* __restrict__ agg, const float* __restrict__ bias,
    const int* __restrict__ batch, float* __restrict__ pooled) {
    __shared__ int s_b[4][16];
    int w    = threadIdx.x >> 5;
    int lane = threadIdx.x & 31;
    int t    = blockIdx.x * 4 + w;
    bool active = (t < NN / 16);   // 50000/16 == 3125 exactly
    int node0 = t * 16;
    if (lane < 16) s_b[w][lane] = active ? batch[node0 + lane] : 0;
    __syncthreads();
    if (!active) return;

    int m  = lane & 15;
    int hi = lane >> 4;
    const _Float16* arow = xh + (size_t)(node0 + m) * DD + hi * 8;
    const _Float16* wrel = Wh + (size_t)RR * (4 * 8 * 512);   // relation 8 = root

    v8f acc[8] = {};
    #pragma unroll
    for (int kt = 0; kt < 4; kt++) {
        v16h a = load_frag(arow + kt * 32, arow + kt * 32 + 16);
        const _Float16* wk = wrel + kt * (8 * 512);
        #pragma unroll
        for (int nt = 0; nt < 8; nt++) {
            const _Float16* bp = wk + nt * 512 + lane * 16;
            v16h b = load_frag(bp, bp + 8);
            acc[nt] = __builtin_amdgcn_wmma_f32_16x16x32_f16(
                false, a, false, b, (short)0, acc[nt], false, false);
        }
    }
    #pragma unroll
    for (int nt = 0; nt < 8; nt++) {
        int col = nt * 16 + m;
        float bcol = bias[col];
        #pragma unroll
        for (int i = 0; i < 8; i++) {
            int row  = i + hi * 8;
            int node = node0 + row;
            float hval = acc[nt][i] + bcol + agg[(size_t)node * DD + col];
            hval = fmaxf(hval, 0.f);
            atomicAdd(&pooled[(size_t)s_b[w][row] * DD + col], hval);
        }
    }
}

// ---------- final classifier: (pooled/gcnt) @ fc_w + fc_b ----------
__global__ void k_final(const float* __restrict__ pooled, const float* __restrict__ gcnt,
                        const float* __restrict__ fc_w, const float* __restrict__ fc_b,
                        float* __restrict__ out) {
    int idx = blockIdx.x * blockDim.x + threadIdx.x;
    if (idx >= GG * CC) return;
    int g = idx >> 4, c = idx & 15;
    float s = 0.f;
    for (int k = 0; k < DD; k++)
        s += pooled[(size_t)g * DD + k] * fc_w[k * CC + c];
    float n = gcnt[g];
    n = n < 1.f ? 1.f : n;
    out[idx] = s / n + fc_b[c];
}

extern "C" void kernel_launch(void* const* d_in, const int* in_sizes, int n_in,
                              void* d_out, int out_size, void* d_ws, size_t ws_size,
                              hipStream_t stream) {
    const float* x      = (const float*)d_in[0];
    const int*   ei     = (const int*)d_in[1];       // [2][E]
    const int*   etype  = (const int*)d_in[2];
    // d_in[3] = edge_attr (unused by reference)
    const int*   batch  = (const int*)d_in[4];
    // d_in[5] = num_graphs scalar (fixed 64)
    const float* basis  = (const float*)d_in[6];
    const float* comp   = (const float*)d_in[7];
    const float* root   = (const float*)d_in[8];
    const float* bias   = (const float*)d_in[9];
    const float* fc_w   = (const float*)d_in[10];
    const float* fc_b   = (const float*)d_in[11];
    float* out = (float*)d_out;

    const int* srcIdx = ei;
    const int* dstIdx = ei + EE;

    // ---- workspace layout (all 256B aligned) ----
    char* ws = (char*)d_ws;
    size_t off = 0;
    float*    agg    = (float*)(ws + off);    off += (size_t)NN * DD * 4;      // 25,600,000
    unsigned* cnt_rd = (unsigned*)(ws + off); off += (size_t)RR * NN * 4;      //  1,600,000
    float*    pooled = (float*)(ws + off);    off += (size_t)GG * DD * 4;      //     32,768
    float*    gcnt   = (float*)(ws + off);    off += 256;                      //        256
    unsigned* ctrl   = (unsigned*)(ws + off); off += 256;                      //        256
    long long zero_words = (long long)(off / 4);
    unsigned* perm = (unsigned*)(ws + off);   off += (size_t)EE * 4;           //  2,400,000
    _Float16* xh   = (_Float16*)(ws + off);   off += (size_t)NN * DD * 2;      // 12,800,000
    _Float16* Wh   = (_Float16*)(ws + off);   off += (size_t)9 * 4 * 8 * 1024; //    294,912

    // 1. zero accumulators/counters
    {
        int blk = 256;
        long long grid = (zero_words + blk - 1) / blk;
        k_zero<<<(unsigned)grid, blk, 0, stream>>>((unsigned*)ws, zero_words);
    }
    // 2. weights -> f16 fragment layout (basis decomposition + root)
    k_prepW<<<(9 * 4 * 8 * 512 + 255) / 256, 256, 0, stream>>>(basis, comp, root, Wh);
    // 3. x -> f16, per-graph node counts
    k_xh<<<((long long)NN * DD + 255) / 256, 256, 0, stream>>>(x, xh, batch, gcnt);
    // 4. edge counts per (relation, dst)
    k_count<<<(EE + 255) / 256, 256, 0, stream>>>(etype, dstIdx, cnt_rd, ctrl);
    // 5. prefix sums
    k_offsets<<<1, 1, 0, stream>>>(ctrl);
    // 6. counting-sort edges by relation
    k_scatter<<<(EE + 255) / 256, 256, 0, stream>>>(etype, perm, ctrl);
    // 7. WMMA edge messages -> agg
    {
        int maxTiles = EE / 16 + RR;               // upper bound on sum ceil(E_r/16)
        int blocks = (maxTiles + 3) / 4;
        k_msg<<<blocks, 128, 0, stream>>>(xh, Wh, srcIdx, dstIdx, perm, cnt_rd, ctrl, agg);
    }
    // 8. WMMA root GEMM + bias + agg + relu + pooling
    {
        int tiles = NN / 16;
        k_h<<<(tiles + 3) / 4, 128, 0, stream>>>(xh, Wh, agg, bias, batch, pooled);
    }
    // 9. classifier head
    k_final<<<(GG * CC + 255) / 256, 256, 0, stream>>>(pooled, gcnt, fc_w, fc_b, out);
    (void)in_sizes; (void)n_in; (void)out_size; (void)ws_size;
}
